// MultiHeadSelfAttention_48773648613414
// MI455X (gfx1250) — matrix operand order repaired and checked by reading, not compiled, not run on hardware
//
#include <hip/hip_runtime.h>

// ---------------------------------------------------------------------------
// MultiHeadSelfAttention on MI455X (gfx1250, wave32, WMMA).
// All three matmul stages run on v_wmma_f32_16x16x32_bf16; LDS tiles are
// double-buffered and filled with GLOBAL_LOAD_ASYNC_TO_LDS_B128 (ASYNCcnt).
// ---------------------------------------------------------------------------

typedef __bf16 bf16_t;
typedef __attribute__((ext_vector_type(16))) __bf16 v16bf;
typedef __attribute__((ext_vector_type(8)))  float  v8f;

union FragU {
  v16bf v;
  uint4 q[2];
  unsigned short h[16];
};

__device__ __forceinline__ bf16_t f32_to_bf16(float f) {
  unsigned u = __builtin_bit_cast(unsigned, f);
  u += 0x7FFFu + ((u >> 16) & 1u);             // round-to-nearest-even
  unsigned short hs = (unsigned short)(u >> 16);
  return __builtin_bit_cast(bf16_t, hs);
}

__device__ __forceinline__ v8f v8f_zero() {
  v8f z = {0.f, 0.f, 0.f, 0.f, 0.f, 0.f, 0.f, 0.f};
  return z;
}

// Raw v_exp_f32: args are <= 0 in online softmax, underflow-to-0 is desired.
__device__ __forceinline__ float fast_exp2(float x) {
  return __builtin_amdgcn_exp2f(x);
}

// Async DMA: global (16B/lane) -> LDS, tracked by ASYNCcnt.
// Generic pointer to __shared__ truncates to the LDS byte offset (ISA 10.2).
__device__ __forceinline__ void async_ld_b128(bf16_t* lds_ptr, const bf16_t* gptr) {
  unsigned loff = (unsigned)(uintptr_t)lds_ptr;
  asm volatile("global_load_async_to_lds_b128 %0, %1, off"
               :: "v"(loff), "v"(gptr)
               : "memory");
}
__device__ __forceinline__ void wait_async0() {
  asm volatile("s_wait_asynccnt 0x0" ::: "memory");
}

// A-operand fragment (16x32 bf16): lane(0..15)=row M,
// halves 0..7 = K koff..koff+7, halves 8..15 = K koff+16..koff+23,
// koff = 0 (lanes 0-15) or 8 (lanes 16-31).
__device__ __forceinline__ v16bf frag_ld_a(const bf16_t* row, int koff) {
  FragU f;
  f.q[0] = *(const uint4*)(row + koff);
  f.q[1] = *(const uint4*)(row + koff + 16);
  return f.v;
}

// B-operand fragment (32x16 bf16): lane(0..15)=col N, halves 0..7 = K koff..koff+7,
// halves 8..15 = K koff+16..koff+23, koff = 0 (lanes 0-15) or 8 (lanes 16-31).
__device__ __forceinline__ v16bf frag_ld_b(const bf16_t* row, int koff) {
  FragU f;
  f.q[0] = *(const uint4*)(row + koff);
  f.q[1] = *(const uint4*)(row + koff + 16);
  return f.v;
}

#define WMMA_BF16(A, B, C) \
  __builtin_amdgcn_wmma_f32_16x16x32_bf16(false, (A), false, (B), (short)0, (C), false, false)

// ---------------------------------------------------------------------------
// fp32 -> bf16 conversion
// ---------------------------------------------------------------------------
__global__ void cvt_f32_bf16(const float* __restrict__ in, bf16_t* __restrict__ out, int n) {
  int i = blockIdx.x * blockDim.x + threadIdx.x;
  if (i < n) out[i] = f32_to_bf16(in[i]);
}

// ---------------------------------------------------------------------------
// GEMM: out[M,Nout] = A[M,K] * W[Nout,K]^T + bias      (torch Linear)
// Block tile 128x128, K-step 32, 256 threads = 8 waves (4x2), each wave does
// a 32x64 strip = 2x4 WMMA tiles.  Double-buffered LDS, async-DMA staged.
// MODE 0: fp32 out.  MODE 1: bf16 Q/K scatter [B,H,N,D] + V scatter [B,H,D,N].
// ---------------------------------------------------------------------------
#define BM 128
#define BN 128
#define BK 32
#define LDT 40   // LDS row stride in bf16 (80 B: 16B-aligned, conflict-free)

template <int MODE>
__global__ __launch_bounds__(256)
void gemm_bf16_wmma(const bf16_t* __restrict__ A, const bf16_t* __restrict__ Bw,
                    const float* __restrict__ bias, int M, int Nout, int K,
                    float* __restrict__ outF,
                    bf16_t* __restrict__ qO, bf16_t* __restrict__ kO, bf16_t* __restrict__ vO) {
  __shared__ __align__(16) bf16_t sA[2 * BM * LDT];
  __shared__ __align__(16) bf16_t sB[2 * BN * LDT];

  const int tid  = threadIdx.x;
  const int lane = tid & 31;
  const int wave = tid >> 5;
  const int wr   = wave >> 1;            // 0..3 : 32-row strip
  const int wc   = wave & 1;             // 0..1 : 64-col strip
  const int bm0  = blockIdx.y * BM;
  const int bn0  = blockIdx.x * BN;
  const int lq   = lane & 15;
  const int koffA = (lane < 16) ? 0 : 8;
  const int koffB = (lane < 16) ? 0 : 8;

  // staging: 128x32 bf16 tile = 512 x 16B chunks, 2 per thread, per matrix
  auto stage = [&](int buf, int k0) {
#pragma unroll
    for (int i = 0; i < 2; ++i) {
      int c  = tid + i * 256;
      int r  = c >> 2;
      int cc = (c & 3) * 8;
      async_ld_b128(sA + buf * (BM * LDT) + r * LDT + cc,
                    A + (size_t)(bm0 + r) * K + k0 + cc);
      async_ld_b128(sB + buf * (BN * LDT) + r * LDT + cc,
                    Bw + (size_t)(bn0 + r) * K + k0 + cc);
    }
  };

  v8f acc[2][4];
#pragma unroll
  for (int i = 0; i < 2; ++i)
#pragma unroll
    for (int j = 0; j < 4; ++j) acc[i][j] = v8f_zero();

  stage(0, 0);
  wait_async0();
  __syncthreads();

  int buf = 0;
  for (int k0 = 0; k0 < K; k0 += BK) {
    if (k0 + BK < K) stage(buf ^ 1, k0 + BK);   // prefetch next tile (async)

    const bf16_t* a0 = sA + buf * (BM * LDT);
    const bf16_t* b0 = sB + buf * (BN * LDT);
    v16bf af[2], bfm[4];
#pragma unroll
    for (int i = 0; i < 2; ++i)
      af[i] = frag_ld_a(a0 + (wr * 32 + i * 16 + lq) * LDT, koffA);
#pragma unroll
    for (int j = 0; j < 4; ++j)
      bfm[j] = frag_ld_b(b0 + (wc * 64 + j * 16 + lq) * LDT, koffB);

#pragma unroll
    for (int i = 0; i < 2; ++i)
#pragma unroll
      for (int j = 0; j < 4; ++j)
        acc[i][j] = WMMA_BF16(af[i], bfm[j], acc[i][j]);

    wait_async0();
    __syncthreads();
    buf ^= 1;
  }

  // C/D layout: vgpr e -> row e (+8 for lanes>=16), lane&15 -> col
  const int rh = (lane & 16) >> 1;   // 0 or 8
#pragma unroll
  for (int i = 0; i < 2; ++i) {
#pragma unroll
    for (int j = 0; j < 4; ++j) {
      int col   = bn0 + wc * 64 + j * 16 + lq;
      float bv  = bias[col];
#pragma unroll
      for (int e = 0; e < 8; ++e) {
        int row   = bm0 + wr * 32 + i * 16 + e + rh;
        float val = acc[i][j][e] + bv;
        if (MODE == 0) {
          outF[(size_t)row * Nout + col] = val;
        } else {
          // col in [0,3072): s*1024 + h*64 + d ; row = b*2048 + n
          int s = col >> 10;
          int h = (col >> 6) & 15;
          int d = col & 63;
          int b = row >> 11;
          int n = row & 2047;
          bf16_t bvv = f32_to_bf16(val);
          if (s == 0)      qO[(((size_t)(b * 16 + h)) * 2048 + n) * 64 + d] = bvv;
          else if (s == 1) kO[(((size_t)(b * 16 + h)) * 2048 + n) * 64 + d] = bvv;
          else             vO[(((size_t)(b * 16 + h)) * 64 + d) * 2048 + n] = bvv; // V^T
        }
      }
    }
  }
}

// ---------------------------------------------------------------------------
// Flash attention.  grid = (N/128, B*H), 256 threads = 8 waves.
// Each wave owns 16 queries.  Computes S^T = K*Q^T so the softmax reduction
// is per-lane (query on lane axis); the exp'd S^T fragments feed the A
// operand of O = P*V with zero cross-lane repacking (C layout == A layout
// half-interleaving per ISA 7.12.2).  V arrives pre-transposed [B,H,D,N].
// Key tile = 64 (16 WMMA per barrier interval); double-buffered async DMA.
// ---------------------------------------------------------------------------
#define AKT  64     // keys per tile
#define SKLD 72     // K / V^T LDS row stride (144 B, 16B-aligned, conflict-free)

__global__ __launch_bounds__(256)
void attn_wmma(const bf16_t* __restrict__ Qm, const bf16_t* __restrict__ Km,
               const bf16_t* __restrict__ Vt, bf16_t* __restrict__ Om) {
  __shared__ __align__(16) bf16_t sK[2 * AKT * SKLD];
  __shared__ __align__(16) bf16_t sVT[2 * 64 * SKLD];

  const int tid  = threadIdx.x;
  const int lane = tid & 31;
  const int wave = tid >> 5;
  const int bh   = blockIdx.y;
  const int b    = bh >> 4;
  const int h    = bh & 15;
  const size_t base = (size_t)bh * (2048 * 64);   // same footprint for K and V^T
  const int q0   = blockIdx.x * 128 + wave * 16;
  const int lq   = lane & 15;
  const int koffA = (lane < 16) ? 0 : 8;
  const int koffB = (lane < 16) ? 0 : 8;
  const float cscale = 0.125f * 1.44269504088896340736f;   // D^-0.5 * log2(e)

  // Each tile: 64x64 bf16 = 512 x 16B chunks per matrix, 2 per thread.
  auto stage = [&](int buf, int kt) {
#pragma unroll
    for (int i = 0; i < 2; ++i) {
      int c  = tid + i * 256;
      int r  = c >> 3;
      int cc = (c & 7) * 8;
      async_ld_b128(sK + buf * (AKT * SKLD) + r * SKLD + cc,
                    Km + base + (size_t)(kt + r) * 64 + cc);        // keys x d
      async_ld_b128(sVT + buf * (64 * SKLD) + r * SKLD + cc,
                    Vt + base + (size_t)r * 2048 + kt + cc);        // d x keys
    }
  };

  // Q^T as B operand: lane = query, d koff..koff+7, koff+16..koff+23 per lane-half.  2 frags (d halves).
  v16bf qf[2];
  qf[0] = frag_ld_b(Qm + base + (size_t)(q0 + lq) * 64,      koffB);
  qf[1] = frag_ld_b(Qm + base + (size_t)(q0 + lq) * 64 + 32, koffB);

  float m_run = -__builtin_inff();
  float l_run = 0.f;
  v8f o[4];
#pragma unroll
  for (int dt = 0; dt < 4; ++dt) o[dt] = v8f_zero();

  stage(0, 0);
  wait_async0();
  __syncthreads();

  int buf = 0;
  for (int kt = 0; kt < 2048; kt += AKT) {
    if (kt + AKT < 2048) stage(buf ^ 1, kt + AKT);   // prefetch next key tile

    const bf16_t* k0p = sK + buf * (AKT * SKLD);
    const bf16_t* v0p = sVT + buf * (64 * SKLD);

    // S^T tiles (keys x queries): vgpr = key, lane = query.  4 key sub-tiles.
    v8f st[4];
#pragma unroll
    for (int mt = 0; mt < 4; ++mt) {
      st[mt] = v8f_zero();
#pragma unroll
      for (int f = 0; f < 2; ++f) {
        v16bf kf = frag_ld_a(k0p + (mt * 16 + lq) * SKLD + f * 32, koffA);
        st[mt] = WMMA_BF16(kf, qf[f], st[mt]);
      }
    }

    // online softmax (scaled log2 domain); lane pair L / L^16 covers all 64 keys
    float mnew = m_run;
    float ss[4][8];
#pragma unroll
    for (int mt = 0; mt < 4; ++mt)
#pragma unroll
      for (int e = 0; e < 8; ++e) {
        ss[mt][e] = st[mt][e] * cscale;
        mnew = fmaxf(mnew, ss[mt][e]);
      }
    mnew = fmaxf(mnew, __shfl_xor(mnew, 16));
    float fac = fast_exp2(m_run - mnew);
    m_run = mnew;

    // P as A operand: pf[kh] covers keys kh*32..kh*32+31
    // (halves 0..7 = sub-tile 2kh, halves 8..15 = sub-tile 2kh+1)
    FragU pf[2];
    float lsum = 0.f;
#pragma unroll
    for (int mt = 0; mt < 4; ++mt)
#pragma unroll
      for (int e = 0; e < 8; ++e) {
        float p = fast_exp2(ss[mt][e] - mnew);
        lsum += p;
        pf[mt >> 1].h[(mt & 1) * 8 + e] = __builtin_bit_cast(unsigned short, f32_to_bf16(p));
      }
    lsum += __shfl_xor(lsum, 16);
    l_run = l_run * fac + lsum;

    // O accumulator rows are queries on the vgpr axis -> broadcast per-row factor
    float face[8];
#pragma unroll
    for (int e = 0; e < 8; ++e)
      face[e] = __shfl(fac, e + ((lane & 16) >> 1));
#pragma unroll
    for (int dt = 0; dt < 4; ++dt)
#pragma unroll
      for (int e = 0; e < 8; ++e) o[dt][e] *= face[e];

    // O += P * V  (B operand = V^T rows: lane = d, contiguous keys)
#pragma unroll
    for (int kh = 0; kh < 2; ++kh)
#pragma unroll
      for (int dt = 0; dt < 4; ++dt) {
        v16bf vf = frag_ld_b(v0p + (dt * 16 + lq) * SKLD + kh * 32, koffB);
        o[dt] = WMMA_BF16(pf[kh].v, vf, o[dt]);
      }

    wait_async0();
    __syncthreads();
    buf ^= 1;
  }

  // normalize and write [B,N,C] bf16 (row = b*2048+n, col = h*64+d)
  float le[8];
#pragma unroll
  for (int e = 0; e < 8; ++e)
    le[e] = 1.0f / __shfl(l_run, e + ((lane & 16) >> 1));
#pragma unroll
  for (int dt = 0; dt < 4; ++dt)
#pragma unroll
    for (int e = 0; e < 8; ++e) {
      int n = q0 + e + ((lane & 16) >> 1);
      int d = dt * 16 + lq;
      Om[(size_t)(b * 2048 + n) * 1024 + h * 64 + d] = f32_to_bf16(o[dt][e] * le[e]);
    }
}

// ---------------------------------------------------------------------------
// Launch
// ---------------------------------------------------------------------------
extern "C" void kernel_launch(void* const* d_in, const int* in_sizes, int n_in,
                              void* d_out, int out_size, void* d_ws, size_t ws_size,
                              hipStream_t stream) {
  (void)in_sizes; (void)n_in; (void)out_size; (void)ws_size;
  const float* x      = (const float*)d_in[0];   // [2,2048,1024]
  const float* w_qkv  = (const float*)d_in[1];   // [3072,1024]
  const float* b_qkv  = (const float*)d_in[2];   // [3072]
  const float* w_proj = (const float*)d_in[3];   // [1024,1024]
  const float* b_proj = (const float*)d_in[4];   // [1024]

  const int M = 4096, C = 1024, C3 = 3072;

  bf16_t* ws = (bf16_t*)d_ws;
  size_t off = 0;
  bf16_t* xb     = ws + off; off += (size_t)M * C;         // 8 MB
  bf16_t* wqkvb  = ws + off; off += (size_t)C3 * C;        // 6 MB
  bf16_t* wprojb = ws + off; off += (size_t)C * C;         // 2 MB
  bf16_t* qb     = ws + off; off += (size_t)M * C;         // 8 MB  [B,H,N,D]
  bf16_t* kb     = ws + off; off += (size_t)M * C;         // 8 MB  [B,H,N,D]
  bf16_t* vtb    = ws + off; off += (size_t)M * C;         // 8 MB  [B,H,D,N]
  bf16_t* ab     = ws + off; off += (size_t)M * C;         // 8 MB  [B,N,C]

  cvt_f32_bf16<<<(M * C + 255) / 256, 256, 0, stream>>>(x, xb, M * C);
  cvt_f32_bf16<<<(C3 * C + 255) / 256, 256, 0, stream>>>(w_qkv, wqkvb, C3 * C);
  cvt_f32_bf16<<<(C * C + 255) / 256, 256, 0, stream>>>(w_proj, wprojb, C * C);

  // qkv = x @ w_qkv^T + b_qkv, scattered to bf16 Q/K [B,H,N,D] and V^T [B,H,D,N]
  gemm_bf16_wmma<1><<<dim3(C3 / BN, M / BM), 256, 0, stream>>>(
      xb, wqkvb, b_qkv, M, C3, C, nullptr, qb, kb, vtb);

  // attention -> bf16 [B,N,C]
  attn_wmma<<<dim3(2048 / 128, 32), 256, 0, stream>>>(qb, kb, vtb, ab);

  // out = attn @ w_proj^T + b_proj  (fp32)
  gemm_bf16_wmma<0><<<dim3(C / BN, M / BM), 256, 0, stream>>>(
      ab, wprojb, b_proj, M, C, C, (float*)d_out, nullptr, nullptr, nullptr);
}
